// NaiveLSTM_46093589021327
// MI455X (gfx1250) — matrix operand order repaired and checked by reading, not compiled, not run on hardware
//
#include <hip/hip_runtime.h>

#define S     1024
#define BATCH 128
#define EMB   512
#define HID   512
#define KIN   1024   // EMB + HID
#define G     2048   // 4*HID

typedef __attribute__((ext_vector_type(16))) __bf16 v16bf;
typedef __attribute__((ext_vector_type(8)))  __bf16 v8bf;
typedef __attribute__((ext_vector_type(8)))  float  v8f;

// ---------------------------------------------------------------------------
// Prep: W [KIN, G] f32 -> transposed bf16 split pair Wt_hi/Wt_lo [G, KIN]
// ---------------------------------------------------------------------------
__global__ __launch_bounds__(256) void lstm_prep_w(const float* __restrict__ W,
                                                   __bf16* __restrict__ Wt_hi,
                                                   __bf16* __restrict__ Wt_lo) {
  int id = blockIdx.x * 256 + threadIdx.x;       // id = k*G + n, coalesced read
  int k = id / G;
  int n = id % G;
  float w  = W[id];
  __bf16 hi = (__bf16)w;
  __bf16 lo = (__bf16)(w - (float)hi);
  Wt_hi[(size_t)n * KIN + k] = hi;
  Wt_lo[(size_t)n * KIN + k] = lo;
}

// ---------------------------------------------------------------------------
// Prep: x [S,B,E] f32 -> bf16 split pair Xhi/Xlo (same layout).  One pass,
// fully parallel; removes ALL conversion VALU from the recurrent loop.
// ---------------------------------------------------------------------------
__global__ __launch_bounds__(256) void lstm_prep_x(const float* __restrict__ x,
                                                   __bf16* __restrict__ Xhi,
                                                   __bf16* __restrict__ Xlo) {
  size_t id = ((size_t)blockIdx.x * 256 + threadIdx.x) * 2;
  float2 a = *(const float2*)(x + id);
  __bf16 h0 = (__bf16)a.x;
  __bf16 h1 = (__bf16)a.y;
  Xhi[id]     = h0;
  Xhi[id + 1] = h1;
  Xlo[id]     = (__bf16)(a.x - (float)h0);
  Xlo[id + 1] = (__bf16)(a.y - (float)h1);
}

__device__ __forceinline__ float fast_sigmoid(float x) {
  return 1.0f / (1.0f + __expf(-x));
}
__device__ __forceinline__ float fast_tanh(float x) {
  float e = __expf(-2.0f * fabsf(x));           // stable for all x
  float r = (1.0f - e) / (1.0f + e);
  return copysignf(r, x);
}

#define FRAG16(P)  __builtin_shufflevector(*(const v8bf*)(P),                  \
                                           *(const v8bf*)((P) + 16),           \
                                           0,1,2,3,4,5,6,7,8,9,10,11,12,13,14,15)

// 4 gate tiles, 3 WMMAs each (split-bf16): z += Ah*Bh + Ah*Bl + Al*Bh
#define GATE(GI, ACC, KROW)                                                   \
    {                                                                         \
      size_t row = (size_t)((GI) * HID + j) * KIN + (size_t)(KROW);           \
      v16bf bh = FRAG16(Wt_hi + row);                                         \
      v16bf bl = FRAG16(Wt_lo + row);                                         \
      ACC = __builtin_amdgcn_wmma_f32_16x16x32_bf16(false, a_hi, false, bh,   \
                                                    (short)0, ACC, false, false); \
      ACC = __builtin_amdgcn_wmma_f32_16x16x32_bf16(false, a_hi, false, bl,   \
                                                    (short)0, ACC, false, false); \
      ACC = __builtin_amdgcn_wmma_f32_16x16x32_bf16(false, a_lo, false, bh,   \
                                                    (short)0, ACC, false, false); \
    }

// ---------------------------------------------------------------------------
// FAST one-timestep kernel.  Grid = 32 WGs x 256 thr (8 waves).
//   workgroup -> one hid 16-col block j0 (all 8 waves read IDENTICAL B
//                fragments -> 8x reuse served by WGP$; no LDS, no barriers)
//   wave      -> one 16-row batch block m0; 4 gate tiles in registers
// A (x|h) arrives pre-split bf16; K-loop is split statically into the x-half
// and h-half so each loop body is branch-free: pure b128 loads + 12 WMMAs.
// ---------------------------------------------------------------------------
__global__ __launch_bounds__(256) void lstm_step_fast(
    const __bf16* __restrict__ Xhi,    // [S, BATCH, EMB]
    const __bf16* __restrict__ Xlo,
    const float*  __restrict__ bias,   // [G]
    const __bf16* __restrict__ Wt_hi,  // [G, KIN]
    const __bf16* __restrict__ Wt_lo,
    const __bf16* __restrict__ hp_hi,  // [BATCH, HID] prev h, split
    const __bf16* __restrict__ hp_lo,
    __bf16*       __restrict__ hn_hi,  // [BATCH, HID] next h, split
    __bf16*       __restrict__ hn_lo,
    float*        __restrict__ cbuf,   // [BATCH, HID] (in/out)
    float*        __restrict__ out,    // [S, BATCH, HID]
    int t)
{
  const int tid  = threadIdx.x;
  const int lane = tid & 31;
  const int wave = tid >> 5;

  const int j0 = blockIdx.x * 16;     // 32 j-blocks, one per workgroup
  const int m0 = wave * 16;           // 8 m-blocks, one per wave

  // 16-bit WMMA lane geometry:
  //   A: lanes 0-15 hold K {0..7,16..23}, lanes 16-31 hold {8..15,24..31}
  //   B: lanes 0-15 hold K {0..7,16..23}, lanes 16-31 hold {8..15,24..31}
  const int rr    = lane & 15;
  const int off   = (lane < 16) ? 0 : 8;
  const int khalf = (lane < 16) ? 0 : 8;
  const int j     = j0 + (lane & 15);

  // prefetch next timestep's pre-split x: 8192 thr x 8 bf16 covers the slice
  if (t + 1 < S) {
    size_t p = (size_t)(t + 1) * BATCH * EMB
             + ((size_t)blockIdx.x * 256 + tid) * 8;
    __builtin_prefetch(Xhi + p, 0, 1);
    __builtin_prefetch(Xlo + p, 0, 1);
  }

  v8f acc0 = {}, acc1 = {}, acc2 = {}, acc3 = {};

  // ---- x-half of K (branch-free loop, invariant base pointers) ----
  {
    const __bf16* pxh = Xhi + (size_t)t * BATCH * EMB
                      + (size_t)(m0 + rr) * EMB + off;
    const __bf16* pxl = Xlo + (size_t)t * BATCH * EMB
                      + (size_t)(m0 + rr) * EMB + off;
    for (int kc = 0; kc < EMB; kc += 32) {
      v16bf a_hi = FRAG16(pxh + kc);
      v16bf a_lo = FRAG16(pxl + kc);
      GATE(0, acc0, kc + khalf)
      GATE(1, acc1, kc + khalf)
      GATE(2, acc2, kc + khalf)
      GATE(3, acc3, kc + khalf)
    }
  }

  // ---- h-half of K (branch-free loop) ----
  {
    const __bf16* phh = hp_hi + (size_t)(m0 + rr) * HID + off;
    const __bf16* phl = hp_lo + (size_t)(m0 + rr) * HID + off;
    for (int kc = 0; kc < HID; kc += 32) {
      v16bf a_hi = FRAG16(phh + kc);
      v16bf a_lo = FRAG16(phl + kc);
      GATE(0, acc0, EMB + kc + khalf)
      GATE(1, acc1, EMB + kc + khalf)
      GATE(2, acc2, EMB + kc + khalf)
      GATE(3, acc3, EMB + kc + khalf)
    }
  }

  // ---- fused gates + state update; h re-split to bf16 at write time ----
  const float bi = bias[j];
  const float bf = bias[HID + j];
  const float bg = bias[2 * HID + j];
  const float bo = bias[3 * HID + j];
  const int mbase = m0 + ((lane < 16) ? 0 : 8);

#pragma unroll
  for (int v = 0; v < 8; ++v) {
    int m = mbase + v;
    float zi = acc0[v] + bi;
    float zf = acc1[v] + bf;
    float zg = acc2[v] + bg;
    float zo = acc3[v] + bo;
    float ig = fast_sigmoid(zi);
    float fg = fast_sigmoid(zf);
    float gg = fast_tanh(zg);
    float og = fast_sigmoid(zo);
    size_t cij = (size_t)m * HID + j;
    float c = fg * cbuf[cij] + ig * gg;
    float h = og * fast_tanh(c);
    cbuf[cij] = c;
    __bf16 hh = (__bf16)h;
    hn_hi[cij] = hh;
    hn_lo[cij] = (__bf16)(h - (float)hh);
    out[((size_t)t * BATCH + m) * HID + j] = h;
  }
}

// ---------------------------------------------------------------------------
// FALLBACK one-timestep kernel (small ws): split A from fp32 on the fly.
// ---------------------------------------------------------------------------
__device__ __forceinline__ void cvt8(float4 a, float4 b, v16bf& hi, v16bf& lo,
                                     int base) {
  float f[8] = {a.x, a.y, a.z, a.w, b.x, b.y, b.z, b.w};
#pragma unroll
  for (int i = 0; i < 8; ++i) {
    __bf16 h = (__bf16)f[i];
    hi[base + i] = h;
    lo[base + i] = (__bf16)(f[i] - (float)h);
  }
}

__global__ __launch_bounds__(256) void lstm_step_conv(
    const float*  __restrict__ x,
    const float*  __restrict__ bias,
    const __bf16* __restrict__ Wt_hi,
    const __bf16* __restrict__ Wt_lo,
    const float*  __restrict__ h_prev,
    float*        __restrict__ h_next,
    float*        __restrict__ cbuf,
    float*        __restrict__ out,
    int t)
{
  const int tid  = threadIdx.x;
  const int lane = tid & 31;
  const int wave = tid >> 5;
  const int j0 = blockIdx.x * 16;
  const int m0 = wave * 16;
  const int rr    = lane & 15;
  const int off   = (lane < 16) ? 0 : 8;
  const int khalf = (lane < 16) ? 0 : 8;
  const int j     = j0 + (lane & 15);

  v8f acc0 = {}, acc1 = {}, acc2 = {}, acc3 = {};

  {
    const float* px = x + (size_t)t * BATCH * EMB + (size_t)(m0 + rr) * EMB + off;
    for (int kc = 0; kc < EMB; kc += 32) {
      const float4* p0 = (const float4*)(px + kc);
      const float4* p1 = (const float4*)(px + kc + 16);
      v16bf a_hi, a_lo;
      cvt8(p0[0], p0[1], a_hi, a_lo, 0);
      cvt8(p1[0], p1[1], a_hi, a_lo, 8);
      GATE(0, acc0, kc + khalf)
      GATE(1, acc1, kc + khalf)
      GATE(2, acc2, kc + khalf)
      GATE(3, acc3, kc + khalf)
    }
  }
  {
    const float* ph = h_prev + (size_t)(m0 + rr) * HID + off;
    for (int kc = 0; kc < HID; kc += 32) {
      const float4* p0 = (const float4*)(ph + kc);
      const float4* p1 = (const float4*)(ph + kc + 16);
      v16bf a_hi, a_lo;
      cvt8(p0[0], p0[1], a_hi, a_lo, 0);
      cvt8(p1[0], p1[1], a_hi, a_lo, 8);
      GATE(0, acc0, EMB + kc + khalf)
      GATE(1, acc1, EMB + kc + khalf)
      GATE(2, acc2, EMB + kc + khalf)
      GATE(3, acc3, EMB + kc + khalf)
    }
  }

  const float bi = bias[j];
  const float bf = bias[HID + j];
  const float bg = bias[2 * HID + j];
  const float bo = bias[3 * HID + j];
  const int mbase = m0 + ((lane < 16) ? 0 : 8);
#pragma unroll
  for (int v = 0; v < 8; ++v) {
    int m = mbase + v;
    float zi = acc0[v] + bi;
    float zf = acc1[v] + bf;
    float zg = acc2[v] + bg;
    float zo = acc3[v] + bo;
    float ig = fast_sigmoid(zi);
    float fg = fast_sigmoid(zf);
    float gg = fast_tanh(zg);
    float og = fast_sigmoid(zo);
    size_t cij = (size_t)m * HID + j;
    float c = fg * cbuf[cij] + ig * gg;
    float h = og * fast_tanh(c);
    cbuf[cij]   = c;
    h_next[cij] = h;
    out[((size_t)t * BATCH + m) * HID + j] = h;
  }
}

// ---------------------------------------------------------------------------
// Host launcher.  All on `stream`, graph-capture safe.
// ws layout: Wt_hi(4Mi) | Wt_lo(4Mi) | cbuf(256Ki) | h fp32 x2 (512Ki) |
//            hs_hi x2, hs_lo x2 (512Ki) | [16Mi:] Xhi(128Mi) | Xlo(128Mi)
// ---------------------------------------------------------------------------
extern "C" void kernel_launch(void* const* d_in, const int* in_sizes, int n_in,
                              void* d_out, int out_size, void* d_ws, size_t ws_size,
                              hipStream_t stream) {
  const float* x = (const float*)d_in[0];   // [S, BATCH, EMB]
  const float* W = (const float*)d_in[1];   // [KIN, G]
  const float* b = (const float*)d_in[2];   // [G]
  float* out = (float*)d_out;               // out | hT | cT

  const size_t MB = 1024ull * 1024ull;
  char* ws = (char*)d_ws;
  __bf16* Wt_hi = (__bf16*)ws;
  __bf16* Wt_lo = (__bf16*)(ws + 4 * MB);
  float*  cbuf  = (float*)(ws + 8 * MB);
  float*  hf0   = cbuf + BATCH * HID;
  float*  hf1   = hf0 + BATCH * HID;
  __bf16* hs_hi0 = (__bf16*)(hf1 + BATCH * HID);
  __bf16* hs_hi1 = hs_hi0 + BATCH * HID;
  __bf16* hs_lo0 = hs_hi1 + BATCH * HID;
  __bf16* hs_lo1 = hs_lo0 + BATCH * HID;
  __bf16* Xhi = (__bf16*)(ws + 16 * MB);
  __bf16* Xlo = (__bf16*)(ws + 144 * MB);
  const size_t NEED_FAST = 272 * MB;

  hipMemsetAsync(cbuf, 0, (size_t)BATCH * HID * sizeof(float), stream);
  lstm_prep_w<<<(KIN * G) / 256, 256, 0, stream>>>(W, Wt_hi, Wt_lo);

  const size_t BH = (size_t)BATCH * HID;

  if (ws_size >= NEED_FAST) {
    hipMemsetAsync(hs_hi0, 0, BH * sizeof(__bf16), stream);
    hipMemsetAsync(hs_lo0, 0, BH * sizeof(__bf16), stream);
    lstm_prep_x<<<((size_t)S * BATCH * EMB) / 512, 256, 0, stream>>>(x, Xhi, Xlo);

    __bf16* hh[2] = {hs_hi0, hs_hi1};
    __bf16* hl[2] = {hs_lo0, hs_lo1};
    for (int t = 0; t < S; ++t) {
      lstm_step_fast<<<32, 256, 0, stream>>>(
          Xhi, Xlo, b, Wt_hi, Wt_lo,
          hh[t & 1], hl[t & 1], hh[(t + 1) & 1], hl[(t + 1) & 1],
          cbuf, out, t);
    }
    // hT == out[S-1]
    hipMemcpyAsync(out + (size_t)S * BH, out + (size_t)(S - 1) * BH,
                   BH * sizeof(float), hipMemcpyDeviceToDevice, stream);
  } else {
    hipMemsetAsync(hf0, 0, BH * sizeof(float), stream);
    float* hb[2] = {hf0, hf1};
    for (int t = 0; t < S; ++t) {
      lstm_step_conv<<<32, 256, 0, stream>>>(x, b, Wt_hi, Wt_lo,
                                             hb[t & 1], hb[(t + 1) & 1],
                                             cbuf, out, t);
    }
    hipMemcpyAsync(out + (size_t)S * BH, hf0,
                   BH * sizeof(float), hipMemcpyDeviceToDevice, stream);
  }

  hipMemcpyAsync(out + (size_t)S * BH + BH, cbuf,
                 BH * sizeof(float), hipMemcpyDeviceToDevice, stream);
}
